// RelativeAttention_62277025792261
// MI455X (gfx1250) — matrix operand order repaired and checked by reading, not compiled, not run on hardware
//
#include <hip/hip_runtime.h>
#include <hip/hip_fp16.h>

typedef __attribute__((ext_vector_type(16))) _Float16 v16h;
typedef __attribute__((ext_vector_type(8)))  float    v8f;

constexpr int cB    = 2;
constexpr int cT    = 2048;
constexpr int cNX   = 512;
constexpr int cH    = 8;
constexpr int cHD   = 64;
constexpr int cNREL = 65;   // 2*KMAX+1 rel table rows
constexpr int cBUCK = 33;   // causal clipped offsets: rid in [0,32]

__device__ __forceinline__ v8f wmma_f16(v16h a, v16h b, v8f c) {
  return __builtin_amdgcn_wmma_f32_16x16x32_f16(false, a, false, b, (short)0, c,
                                                false, false);
}

// A-matrix (16x32 f16, MxK) fragment in this code's contiguous k order:
// lane M = lane&15; VGPR r holds K = hi*16 + r*2, +1
__device__ __forceinline__ v16h frag_a_f16(const _Float16* rowptr, int hi) {
  v16h a;
#pragma unroll
  for (int r = 0; r < 8; ++r) {
    const int k = (r << 1) + (hi << 4);
    a[2 * r]     = rowptr[k];
    a[2 * r + 1] = rowptr[k + 1];
  }
  return a;
}

__device__ __forceinline__ v16h frag_a_f32(const float* rowptr, int hi) {
  v16h a;
#pragma unroll
  for (int r = 0; r < 8; ++r) {
    const int k = (r << 1) + (hi << 4);
    a[2 * r]     = (_Float16)rowptr[k];
    a[2 * r + 1] = (_Float16)rowptr[k + 1];
  }
  return a;
}

// ---------------------------------------------------------------------------
// Generic 128x64-tile WMMA GEMM, 8 waves/block, each wave owns a 16x64 strip.
// MODE 0: A=f32 (x), epilogue scatters into q(scaled f16)/k(f16)/vT(f16).
// MODE 1: A=f16 (attn out), epilogue writes f32 C (+bias).
// ---------------------------------------------------------------------------
template <int MODE>
__global__ __launch_bounds__(256) void gemm_wmma_kernel(
    const float* __restrict__ Af32, const _Float16* __restrict__ Af16,
    const float* __restrict__ Bmat, const float* __restrict__ bias,
    _Float16* __restrict__ q_out, _Float16* __restrict__ k_out,
    _Float16* __restrict__ vT_out, float* __restrict__ c_out,
    int M, int N, int K) {
  __shared__ _Float16 Bt[64 * 32];  // B tile transposed: [n within tile][k]

  const int tid  = threadIdx.x;
  const int lane = tid & 31;
  const int wave = tid >> 5;
  const int hi   = lane >> 4;
  const int ln   = lane & 15;

  const int mbase = blockIdx.x * 128 + wave * 16;
  const int nbase = blockIdx.y * 64;

  const v8f zero8 = {0.f, 0.f, 0.f, 0.f, 0.f, 0.f, 0.f, 0.f};
  v8f c[4] = {zero8, zero8, zero8, zero8};

  const int arow = mbase + ln;

  for (int k0 = 0; k0 < K; k0 += 32) {
    if (k0) __syncthreads();
    // stage B chunk [32 x 64] f32 -> LDS f16 transposed (coalesced along n)
    for (int e = tid; e < 32 * 64; e += 256) {
      const int kk = e >> 6, nn = e & 63;
      Bt[nn * 32 + kk] = (_Float16)Bmat[(size_t)(k0 + kk) * N + nbase + nn];
    }
    __syncthreads();

    v16h a;
    if (MODE == 0) a = frag_a_f32(Af32 + (size_t)arow * K + k0, hi);
    else           a = frag_a_f16(Af16 + (size_t)arow * K + k0, hi);

#pragma unroll
    for (int fj = 0; fj < 4; ++fj) {
      // B frag: lane n = fj*16+ln, K = hi*16 + 0..15 contiguous
      v16h bf = *(const v16h*)&Bt[(fj * 16 + ln) * 32 + hi * 16];
      c[fj] = wmma_f16(a, bf, c[fj]);
    }
  }

#pragma unroll
  for (int fj = 0; fj < 4; ++fj) {
#pragma unroll
    for (int r = 0; r < 8; ++r) {
      const int m = mbase + r + hi * 8;  // C layout: M = r + hi*8
      const int n = nbase + fj * 16 + ln;
      const float val = c[fj][r] + bias[n];
      if (MODE == 0) {
        const int which = n >> 9;        // 0:q 1:k 2:v
        const int nn = n & 511;
        const int hh = nn >> 6, d = nn & 63;
        const int bb = m >> 11, t = m & (cT - 1);
        const size_t bhrow = (size_t)(bb * cH + hh);
        if (which == 0)
          q_out[(bhrow * cT + t) * cHD + d] = (_Float16)(val * 0.125f);  // 1/sqrt(64)
        else if (which == 1)
          k_out[(bhrow * cT + t) * cHD + d] = (_Float16)val;
        else
          vT_out[(bhrow * cHD + d) * cT + t] = (_Float16)val;  // v transposed [HD,T]
      } else {
        c_out[(size_t)m * N + n] = val;
      }
    }
  }
}

// ---------------------------------------------------------------------------
// qrel[bh,t,r] = q_scaled[bh,t,:] . rel_keys_tab[r,:]   (65 distinct columns)
// ---------------------------------------------------------------------------
__global__ __launch_bounds__(256) void qrel_kernel(
    const _Float16* __restrict__ q, const float* __restrict__ relk,
    float* __restrict__ qrel, int total) {
  const int idx = blockIdx.x * 256 + threadIdx.x;
  if (idx >= total) return;
  const int r  = idx % cNREL;
  const int bt = idx / cNREL;
  const _Float16* qp = q + (size_t)bt * cHD;
  const float* kp = relk + (size_t)r * cHD;
  float s = 0.f;
#pragma unroll
  for (int d = 0; d < cHD; ++d) s += (float)qp[d] * kp[d];
  qrel[idx] = s;
}

// ---------------------------------------------------------------------------
// Fused flash attention with relative-position terms.
// grid = (T/128, B*H); block = 256 (8 waves); each wave owns 16 rows,
// iterates 64-column key tiles up to its causal bound. No block barriers.
// ---------------------------------------------------------------------------
__global__ __launch_bounds__(256) void attn_kernel(
    const _Float16* __restrict__ qh, const _Float16* __restrict__ kh,
    const _Float16* __restrict__ vT, const float* __restrict__ qrel,
    const int* __restrict__ rel, const float* __restrict__ relw,
    const float* __restrict__ relv, _Float16* __restrict__ attnout) {
  __shared__ _Float16 Pl[8][16 * 64];          // per-wave P staging (C->A relayout)
  __shared__ float    Bk[8][16][cBUCK + 1];    // per-wave rel-value bucket masses
  __shared__ float    Scl[8][16];              // per-row rescale broadcast

  const int tid  = threadIdx.x;
  const int lane = tid & 31;
  const int wave = tid >> 5;
  const int hi   = lane >> 4;
  const int ln   = lane & 15;

  const int bh = blockIdx.y;
  const int b  = bh >> 3;
  const int h  = bh & 7;
  const int i0 = blockIdx.x * 128 + wave * 16;

  const _Float16* qb = qh + (size_t)bh * cT * cHD;
  const _Float16* kb = kh + (size_t)bh * cT * cHD;
  const _Float16* vb = vT + (size_t)bh * cHD * cT;
  const float* qrelb = qrel + (size_t)bh * cT * cNREL;
  const int*   relb  = rel + (size_t)b * cT * cT;

  // q A-fragments for this wave's 16 rows (K=HD=64 in two chunks of 32)
  const _Float16* qrow = qb + (size_t)(i0 + ln) * cHD;
  const v16h aq0 = frag_a_f16(qrow, hi);
  const v16h aq1 = frag_a_f16(qrow + 32, hi);

  const v8f zero8 = {0.f, 0.f, 0.f, 0.f, 0.f, 0.f, 0.f, 0.f};
  v8f o[4] = {zero8, zero8, zero8, zero8};
  float mrow[8], lrow[8];
#pragma unroll
  for (int r = 0; r < 8; ++r) { mrow[r] = -1.0e30f; lrow[r] = 0.f; }

  // zero this wave's bucket accumulators (wave-private)
  float* bkw = &Bk[wave][0][0];
  for (int e = lane; e < 16 * (cBUCK + 1); e += 32) bkw[e] = 0.f;
  asm volatile("s_wait_dscnt 0x0" ::: "memory");

  const int jtiles = ((i0 + 15) >> 6) + 1;  // causal bound for this wave
  for (int jt = 0; jt < jtiles; ++jt) {
    const int jbase = jt << 6;

    // ---- S = q_scaled @ k^T  (16 x 64) ----
    v8f s[4];
#pragma unroll
    for (int fj = 0; fj < 4; ++fj) {
      const _Float16* kp = kb + (size_t)(jbase + fj * 16 + ln) * cHD + hi * 16;
      v16h b0 = *(const v16h*)kp;
      v16h b1 = *(const v16h*)(kp + 32);
      v8f acc = zero8;
      acc = wmma_f16(aq0, b0, acc);
      acc = wmma_f16(aq1, b1, acc);
      s[fj] = acc;
    }

    // ---- add gathered relative-key term + tree bias; causal mask; tile max ----
    float tmax[8];
#pragma unroll
    for (int r = 0; r < 8; ++r) tmax[r] = -1.0e30f;
#pragma unroll
    for (int fj = 0; fj < 4; ++fj) {
#pragma unroll
      for (int r = 0; r < 8; ++r) {
        const int i = i0 + r + hi * 8;
        const int j = jbase + fj * 16 + ln;
        float val = -1.0e30f;
        if (j <= i) {
          const int dj  = j - i;                         // <= 0
          const int rid = (dj < -32) ? 0 : (dj + 32);    // causal clip -> [0,32]
          val = s[fj][r] + qrelb[(size_t)i * cNREL + rid] +
                relw[(size_t)relb[(size_t)i * cT + j] * cH + h];
        }
        s[fj][r] = val;
        tmax[r]  = fmaxf(tmax[r], val);
      }
    }
    // row max across the 16 lanes of this half-wave
#pragma unroll
    for (int r = 0; r < 8; ++r) {
      float v = tmax[r];
#pragma unroll
      for (int off = 1; off < 16; off <<= 1) v = fmaxf(v, __shfl_xor(v, off, 32));
      tmax[r] = v;
    }

    // ---- online-softmax rescale of O, l, and bucket masses ----
    float scl[8];
#pragma unroll
    for (int r = 0; r < 8; ++r) {
      const float mnew = fmaxf(mrow[r], tmax[r]);
      const float sc   = __expf(mrow[r] - mnew);
      mrow[r] = mnew;
      lrow[r] *= sc;
      scl[r] = sc;
#pragma unroll
      for (int fo = 0; fo < 4; ++fo) o[fo][r] *= sc;
    }
    if (ln == 0) {
#pragma unroll
      for (int r = 0; r < 8; ++r) Scl[wave][r + hi * 8] = scl[r];
    }
    asm volatile("s_wait_dscnt 0x0" ::: "memory");
    for (int e = lane; e < 16 * cBUCK; e += 32) {
      const int rr = e / cBUCK, cc = e % cBUCK;
      Bk[wave][rr][cc] *= Scl[wave][rr];
    }

    // ---- exponentiate; row sums; bucket ds_add_f32; stage P for relayout ----
    float lsum[8];
#pragma unroll
    for (int r = 0; r < 8; ++r) lsum[r] = 0.f;
#pragma unroll
    for (int fj = 0; fj < 4; ++fj) {
#pragma unroll
      for (int r = 0; r < 8; ++r) {
        const int i = i0 + r + hi * 8;
        const int j = jbase + fj * 16 + ln;
        float p = 0.f;
        if (j <= i) {
          p = __expf(s[fj][r] - mrow[r]);
          const int dj  = j - i;
          const int rid = (dj < -32) ? 0 : (dj + 32);
          atomicAdd(&Bk[wave][r + hi * 8][rid], p);
        }
        lsum[r] += p;
        Pl[wave][(r + hi * 8) * 64 + fj * 16 + ln] = (_Float16)p;
      }
    }
#pragma unroll
    for (int r = 0; r < 8; ++r) {
      float v = lsum[r];
#pragma unroll
      for (int off = 1; off < 16; off <<= 1) v += __shfl_xor(v, off, 32);
      lrow[r] += v;
    }
    asm volatile("s_wait_dscnt 0x0" ::: "memory");

    // ---- O += P @ V  (vT is [HD,T] so B-frags are contiguous 32B loads) ----
    const v16h pa0 = frag_a_f16(&Pl[wave][ln * 64], hi);
    const v16h pa1 = frag_a_f16(&Pl[wave][ln * 64 + 32], hi);
#pragma unroll
    for (int fo = 0; fo < 4; ++fo) {
      const _Float16* vp = vb + (size_t)(fo * 16 + ln) * cT + jbase + hi * 16;
      v16h b0 = *(const v16h*)vp;
      v16h b1 = *(const v16h*)(vp + 32);
      o[fo] = wmma_f16(pa0, b0, o[fo]);
      o[fo] = wmma_f16(pa1, b1, o[fo]);
    }
  }

  asm volatile("s_wait_dscnt 0x0" ::: "memory");
  // ---- finalize: add bucketed relative-value term, normalize, merge heads ----
#pragma unroll
  for (int fo = 0; fo < 4; ++fo) {
#pragma unroll
    for (int r = 0; r < 8; ++r) {
      const int i = i0 + r + hi * 8;
      const int d = fo * 16 + ln;
      float acc = o[fo][r];
      for (int rid = 0; rid < cBUCK; ++rid)
        acc += Bk[wave][r + hi * 8][rid] * relv[rid * cHD + d];
      acc /= lrow[r];
      attnout[(size_t)(b * cT + i) * cNX + h * cHD + d] = (_Float16)acc;
    }
  }
}

// ---------------------------------------------------------------------------
extern "C" void kernel_launch(void* const* d_in, const int* in_sizes, int n_in,
                              void* d_out, int out_size, void* d_ws,
                              size_t ws_size, hipStream_t stream) {
  (void)in_sizes; (void)n_in; (void)out_size; (void)ws_size;
  const float* x      = (const float*)d_in[0];
  const int*   rel    = (const int*)d_in[1];
  const float* W_attn = (const float*)d_in[2];
  const float* b_attn = (const float*)d_in[3];
  const float* W_proj = (const float*)d_in[4];
  const float* b_proj = (const float*)d_in[5];
  const float* relw   = (const float*)d_in[6];
  const float* relk   = (const float*)d_in[7];
  const float* relv   = (const float*)d_in[8];
  float* out = (float*)d_out;

  // workspace carve-up (~25 MB)
  const size_t headHalves = (size_t)cB * cH * cT * cHD;  // 2,097,152 halves
  _Float16* qh  = (_Float16*)d_ws;
  _Float16* kh  = qh + headHalves;
  _Float16* vT  = kh + headHalves;
  _Float16* att = vT + headHalves;                       // [B*T, NX] f16
  float* qrel = (float*)(att + (size_t)cB * cT * cNX);   // [B*H*T, 65] f32

  const int Mrows = cB * cT;  // 4096

  // 1) QKV projection; epilogue scatters q(scaled)/k/v(transposed) as f16
  gemm_wmma_kernel<0><<<dim3(Mrows / 128, (3 * cNX) / 64), 256, 0, stream>>>(
      x, nullptr, W_attn, b_attn, qh, kh, vT, nullptr, Mrows, 3 * cNX, cNX);

  // 2) q @ rel_keys^T -> [B*H*T, 65]
  {
    const int total = cB * cH * cT * cNREL;
    qrel_kernel<<<(total + 255) / 256, 256, 0, stream>>>(qh, relk, qrel, total);
  }

  // 3) fused flash attention with relative terms
  attn_kernel<<<dim3(cT / 128, cB * cH), 256, 0, stream>>>(
      qh, kh, vT, qrel, rel, relw, relv, att);

  // 4) output projection -> f32 result
  gemm_wmma_kernel<1><<<dim3(Mrows / 128, cNX / 64), 256, 0, stream>>>(
      nullptr, att, W_proj, b_proj, nullptr, nullptr, nullptr, out, Mrows, cNX,
      cNX);
}
